// MultiHeadedAttention_11304353923231
// MI455X (gfx1250) — matrix operand order repaired and checked by reading, not compiled, not run on hardware
//
#include <hip/hip_runtime.h>
#include <hip/hip_bf16.h>

#define EMBED 1024
#define HEADS 16
#define HEAD_DIM 64
#define BATCH 2
#define SEQ 2048

typedef __attribute__((ext_vector_type(16))) _Float16 v16h;
typedef __attribute__((ext_vector_type(8)))  _Float16 v8h;
typedef __attribute__((ext_vector_type(8)))  float    v8f;
typedef __attribute__((ext_vector_type(4)))  int      v4i;

union F16x16 { v16h v; v8h h[2]; };

// ---- async global->LDS copy (gfx1250 path, probed) --------------------------
#if defined(__has_builtin)
#if __has_builtin(__builtin_amdgcn_global_load_async_to_lds_b128)
#define HAVE_ASYNC_LDS 1
#endif
#endif

#ifdef HAVE_ASYNC_LDS
typedef __attribute__((address_space(1))) v4i* g_v4i_ptr;  // global v4i*
typedef __attribute__((address_space(3))) v4i* l_v4i_ptr;  // LDS v4i*
#endif

__device__ __forceinline__ void copy16_to_lds(_Float16* lds_dst, const _Float16* gsrc) {
#ifdef HAVE_ASYNC_LDS
    __builtin_amdgcn_global_load_async_to_lds_b128(
        (g_v4i_ptr)(void*)gsrc, (l_v4i_ptr)(void*)lds_dst, 0, 0);
#else
    *(v8h*)lds_dst = *(const v8h*)gsrc;
#endif
}

__device__ __forceinline__ void wait_async_le2() {
#ifdef HAVE_ASYNC_LDS
#if __has_builtin(__builtin_amdgcn_s_wait_asynccnt)
    __builtin_amdgcn_s_wait_asynccnt(2);
#else
    asm volatile("s_wait_asynccnt 0x2" ::: "memory");
#endif
#endif
}
__device__ __forceinline__ void wait_async_le0() {
#ifdef HAVE_ASYNC_LDS
#if __has_builtin(__builtin_amdgcn_s_wait_asynccnt)
    __builtin_amdgcn_s_wait_asynccnt(0);
#else
    asm volatile("s_wait_asynccnt 0x0" ::: "memory");
#endif
#endif
}

// ---- wave32 XOR lane exchange via ds_swizzle (imm pattern, no index VGPRs) --
#define SWZ_XOR(v, imm) \
    __int_as_float(__builtin_amdgcn_ds_swizzle(__float_as_int(v), (imm)))
__device__ __forceinline__ float red_max16(float t) {
    t = fmaxf(t, SWZ_XOR(t, 0x041f)); // xor 1
    t = fmaxf(t, SWZ_XOR(t, 0x081f)); // xor 2
    t = fmaxf(t, SWZ_XOR(t, 0x101f)); // xor 4
    t = fmaxf(t, SWZ_XOR(t, 0x201f)); // xor 8
    return t;
}
__device__ __forceinline__ float red_sum16(float t) {
    t += SWZ_XOR(t, 0x041f);
    t += SWZ_XOR(t, 0x081f);
    t += SWZ_XOR(t, 0x101f);
    t += SWZ_XOR(t, 0x201f);
    return t;
}

// ---- WMMA helpers -----------------------------------------------------------
// A-matrix 16x32 f16 fragment (lane&15 = row M; lane<16 holds K 0..7 & 16..23,
// lane>=16 holds K 8..15 & 24..31). `p` points at this row's K=0 element.
__device__ __forceinline__ v16h load_a32(const _Float16* p, int lane) {
    const int klo = (lane & 16) ? 8 : 0;
    F16x16 f;
    f.h[0] = *(const v8h*)(p + klo);
    f.h[1] = *(const v8h*)(p + klo + 16);
    return f.v;
}
// B-matrix 32x16 f16 fragment (lane&15 = column N; lane<16 holds K 0..7 & 16..23,
// lane>=16 holds K 8..15 & 24..31). `p` points at this column's K=0 element, with K
// contiguous in memory.
__device__ __forceinline__ v16h load_b32(const _Float16* p, int lane) {
    const int klo = (lane & 16) ? 8 : 0;
    F16x16 f;
    f.h[0] = *(const v8h*)(p + klo);
    f.h[1] = *(const v8h*)(p + klo + 16);
    return f.v;
}
__device__ __forceinline__ v8f wmma_f16(v16h a, v16h b, v8f c) {
    return __builtin_amdgcn_wmma_f32_16x16x32_f16(
        /*neg_a=*/false, a, /*neg_b=*/false, b,
        /*c_mod=*/(short)0, c, /*reuse_a=*/false, /*reuse_b=*/false);
}

// ---- fp32 -> fp16 conversion ------------------------------------------------
__global__ void __launch_bounds__(256)
cvt_f32_f16(const float* __restrict__ src, _Float16* __restrict__ dst, int n) {
    int i = blockIdx.x * blockDim.x + threadIdx.x;
    if (i < n) dst[i] = (_Float16)src[i];
}

// ---- fc1 GEMM: out[m][n] = sum_k X16[m][k] * W16[n][k] + bias[n] ------------
// M = 4096 (B*L), N = K = 1024.  Block = 8 waves, wave tile = 32x32 (4 WMMA acc).
enum { MODE_F16 = 0, MODE_F16T = 1, MODE_F32 = 2 };

template <int MODE>
__global__ void __launch_bounds__(256)
fc1_gemm(const _Float16* __restrict__ X, const _Float16* __restrict__ W,
         const float* __restrict__ bias,
         _Float16* __restrict__ out16, float* __restrict__ out32) {
    const int lane = threadIdx.x & 31;
    const int wave = threadIdx.x >> 5;
    const int M0 = blockIdx.x * 64 + (wave & 1) * 32;
    const int N0 = blockIdx.y * 128 + (wave >> 1) * 32;
    const int row = lane & 15;

    v8f acc00 = {}, acc01 = {}, acc10 = {}, acc11 = {};
    const _Float16* xr0 = X + (size_t)(M0 + row) * EMBED;
    const _Float16* xr1 = X + (size_t)(M0 + 16 + row) * EMBED;
    const _Float16* wr0 = W + (size_t)(N0 + row) * EMBED;
    const _Float16* wr1 = W + (size_t)(N0 + 16 + row) * EMBED;

    for (int k0 = 0; k0 < EMBED; k0 += 32) {
        v16h a0 = load_a32(xr0 + k0, lane);
        v16h a1 = load_a32(xr1 + k0, lane);
        v16h b0 = load_b32(wr0 + k0, lane);
        v16h b1 = load_b32(wr1 + k0, lane);
        acc00 = wmma_f16(a0, b0, acc00);
        acc01 = wmma_f16(a0, b1, acc01);
        acc10 = wmma_f16(a1, b0, acc10);
        acc11 = wmma_f16(a1, b1, acc11);
    }

    const int mhi = (lane & 16) ? 8 : 0;
#pragma unroll
    for (int mt = 0; mt < 2; ++mt) {
#pragma unroll
        for (int nt = 0; nt < 2; ++nt) {
            v8f a = mt ? (nt ? acc11 : acc10) : (nt ? acc01 : acc00);
            const int n = N0 + nt * 16 + row;
            const float bv = bias[n];
#pragma unroll
            for (int i = 0; i < 8; ++i) {
                const int m = M0 + mt * 16 + i + mhi;
                const float v = a[i] + bv;
                if (MODE == MODE_F32) {
                    out32[(size_t)m * EMBED + n] = v;
                } else if (MODE == MODE_F16) {
                    out16[(size_t)m * EMBED + n] = (_Float16)v;
                } else { // MODE_F16T: [B][E][L] so P*V B-fragments are contiguous
                    const int b = m >> 11, l = m & (SEQ - 1);
                    out16[(size_t)b * EMBED * SEQ + (size_t)n * SEQ + l] = (_Float16)v;
                }
            }
        }
    }
}

// ---- Flash attention --------------------------------------------------------
// Block = 8 waves, all on the same (b,h); each wave owns 16 query rows.
// K/V tiles (32 keys) are staged once per block into double-buffered LDS via
// the async global->LDS path, instead of 8x redundant per-wave global loads.
__global__ void __launch_bounds__(256)
attention_kernel(const _Float16* __restrict__ Q, const _Float16* __restrict__ K,
                 const _Float16* __restrict__ Vt, _Float16* __restrict__ ctx) {
    __shared__ __align__(16) _Float16 ldsK[2][32 * 64]; // [key][d]   4KB each
    __shared__ __align__(16) _Float16 ldsV[2][64 * 32]; // [d][key]   4KB each
    __shared__ __align__(16) _Float16 plds[8][16 * 32]; // P tile per wave

    const int tid = threadIdx.x;
    const int lane = tid & 31;
    const int wave = tid >> 5;
    const int h = blockIdx.y;
    const int b = blockIdx.z;
    const int q0 = (blockIdx.x * 8 + wave) * 16;
    const int row = lane & 15;
    const int mhi = (lane & 16) ? 8 : 0;

    const size_t qkBase = (size_t)b * SEQ * EMBED + (size_t)h * HEAD_DIM;
    const size_t vBase  = (size_t)b * EMBED * SEQ + (size_t)(h * HEAD_DIM) * SEQ;

    // cooperative K/V tile stage: K = 32 rows x 128B, V = 64 rows x 64B
    auto stageKV = [&](int kt, int bb) {
        {
            const int r = tid >> 3, c = (tid & 7) * 8;
            copy16_to_lds(&ldsK[bb][r * 64 + c],
                          K + qkBase + (size_t)(kt + r) * EMBED + c);
        }
        {
            const int r = tid >> 2, c = (tid & 3) * 8;
            copy16_to_lds(&ldsV[bb][r * 32 + c],
                          Vt + vBase + (size_t)r * SEQ + kt + c);
        }
    };

    // Q fragments for d=0..31 and d=32..63 (held for the whole key loop)
    const _Float16* qrow = Q + qkBase + (size_t)(q0 + row) * EMBED;
    const v16h qa0 = load_a32(qrow, lane);
    const v16h qa1 = load_a32(qrow + 32, lane);

    float mrun[8], lrun[8];
#pragma unroll
    for (int i = 0; i < 8; ++i) { mrun[i] = -3.0e38f; lrun[i] = 0.0f; }
    v8f acc0 = {}, acc1 = {}, acc2 = {}, acc3 = {};

    _Float16* pbuf = plds[wave];
    const float scale = 0.03125f; // 1/sqrt(EMBED) = 1/32 (faithful to reference)

    stageKV(0, 0);
    for (int kt = 0; kt < SEQ; kt += 32) {
        const int cur = (kt >> 5) & 1;
        const bool more = (kt + 32) < SEQ;
        if (more) {
            stageKV(kt + 32, cur ^ 1);
            wait_async_le2(); // cur buffer's 2 async ops (this wave) complete
        } else {
            wait_async_le0();
        }
        __syncthreads();

        // ---- S = Q * K^T (two 16x16 f32 tiles: keys +0..15, +16..31)
        const _Float16* kb0 = &ldsK[cur][(size_t)row * 64];
        const _Float16* kb1 = &ldsK[cur][(size_t)(16 + row) * 64];
        v8f s0 = {}, s1 = {};
        s0 = wmma_f16(qa0, load_b32(kb0, lane), s0);
        s0 = wmma_f16(qa1, load_b32(kb0 + 32, lane), s0);
        s1 = wmma_f16(qa0, load_b32(kb1, lane), s1);
        s1 = wmma_f16(qa1, load_b32(kb1 + 32, lane), s1);

        // ---- online softmax (row stats replicated per 16-lane half)
#pragma unroll
        for (int i = 0; i < 8; ++i) {
            const float a = s0[i] * scale;
            const float c = s1[i] * scale;
            const float t = red_max16(fmaxf(a, c));
            const float mnew = fmaxf(mrun[i], t);
            const float al = __expf(mrun[i] - mnew);
            const float p0 = __expf(a - mnew);
            const float p1 = __expf(c - mnew);
            const float rs = red_sum16(p0 + p1);
            lrun[i] = lrun[i] * al + rs;
            mrun[i] = mnew;
            acc0[i] *= al; acc1[i] *= al; acc2[i] *= al; acc3[i] *= al;
            // stage P (16x32 f16, row-major) for the A-fragment reload
            const int m = i + mhi;
            pbuf[m * 32 + row]      = (_Float16)p0;
            pbuf[m * 32 + 16 + row] = (_Float16)p1;
        }
        asm volatile("s_wait_dscnt 0x0" ::: "memory");

        // reload P with the A-matrix striping (cross-lane redistribution)
        v16h pa;
        {
            const int klo = (lane & 16) ? 8 : 0;
            F16x16 f;
            f.h[0] = *(const v8h*)(pbuf + row * 32 + klo);
            f.h[1] = *(const v8h*)(pbuf + row * 32 + klo + 16);
            pa = f.v;
        }

        // ---- ctx += P * V  (ldsV row d holds 32 contiguous keys)
        acc0 = wmma_f16(pa, load_b32(&ldsV[cur][(size_t)row * 32], lane), acc0);
        acc1 = wmma_f16(pa, load_b32(&ldsV[cur][(size_t)(16 + row) * 32], lane), acc1);
        acc2 = wmma_f16(pa, load_b32(&ldsV[cur][(size_t)(32 + row) * 32], lane), acc2);
        acc3 = wmma_f16(pa, load_b32(&ldsV[cur][(size_t)(48 + row) * 32], lane), acc3);

        __syncthreads(); // reads of `cur` done before it is overwritten next+1
    }

    // ---- normalize + store ctx (fp16, [B][L][E] with heads interleaved)
#pragma unroll
    for (int dt = 0; dt < 4; ++dt) {
        v8f a = dt == 0 ? acc0 : dt == 1 ? acc1 : dt == 2 ? acc2 : acc3;
        const int d = dt * 16 + row;
#pragma unroll
        for (int i = 0; i < 8; ++i) {
            const int m = q0 + i + mhi;
            const float v = a[i] / lrun[i];
            ctx[qkBase + (size_t)m * EMBED + d] = (_Float16)v;
        }
    }
}

// ---- host-side launch -------------------------------------------------------
extern "C" void kernel_launch(void* const* d_in, const int* in_sizes, int n_in,
                              void* d_out, int out_size, void* d_ws, size_t ws_size,
                              hipStream_t stream) {
    const float* query  = (const float*)d_in[0];
    const float* key    = (const float*)d_in[1];
    const float* values = (const float*)d_in[2];
    const float* W1     = (const float*)d_in[3];
    const float* b1     = (const float*)d_in[4];
    float* out = (float*)d_out;

    const size_t SZ_W = (size_t)EMBED * EMBED;        // 1,048,576
    const size_t SZ_X = (size_t)BATCH * SEQ * EMBED;  // 4,194,304

    _Float16* W16  = (_Float16*)d_ws;
    _Float16* Xq   = W16 + SZ_W;
    _Float16* Xk   = Xq + SZ_X;
    _Float16* Xv   = Xk + SZ_X;
    _Float16* Q16  = Xv + SZ_X;
    _Float16* K16  = Q16 + SZ_X;
    _Float16* Vt16 = K16 + SZ_X;
    _Float16* ctx  = Xq;  // reuse: query fp16 is dead after the Q projection

    const int T = 256;
    cvt_f32_f16<<<(int)((SZ_W + T - 1) / T), T, 0, stream>>>(W1, W16, (int)SZ_W);
    cvt_f32_f16<<<(int)((SZ_X + T - 1) / T), T, 0, stream>>>(query, Xq, (int)SZ_X);
    cvt_f32_f16<<<(int)((SZ_X + T - 1) / T), T, 0, stream>>>(key, Xk, (int)SZ_X);
    cvt_f32_f16<<<(int)((SZ_X + T - 1) / T), T, 0, stream>>>(values, Xv, (int)SZ_X);

    dim3 gemm_grid((BATCH * SEQ) / 64, EMBED / 128);
    fc1_gemm<MODE_F16><<<gemm_grid, 256, 0, stream>>>(Xq, W16, b1, Q16, nullptr);
    fc1_gemm<MODE_F16><<<gemm_grid, 256, 0, stream>>>(Xk, W16, b1, K16, nullptr);
    fc1_gemm<MODE_F16T><<<gemm_grid, 256, 0, stream>>>(Xv, W16, b1, Vt16, nullptr);

    dim3 attn_grid(SEQ / (16 * 8), HEADS, BATCH);
    attention_kernel<<<attn_grid, 256, 0, stream>>>(Q16, K16, Vt16, ctx);

    fc1_gemm<MODE_F32><<<gemm_grid, 256, 0, stream>>>(ctx, W16, b1, nullptr, out);
}
